// OutputBlock_47115791237973
// MI455X (gfx1250) — matrix operand order repaired and checked by reading, not compiled, not run on hardware
//
#include <hip/hip_runtime.h>
#include <hip/hip_bf16.h>
#include <math.h>

#define H      128
#define OUT_F  128
#define NRAD   6
#define NEXP   8
#define NBKT   9           // 8 experts + 1 passthrough bucket
#define SEL_LO 0.8f
#define SEL_DIV 5.0001f
#define HP     136         // padded LDS row stride (bf16 elems); 272B = 16B-aligned rows

typedef __bf16 bf16_t;
typedef __attribute__((ext_vector_type(16))) __bf16 v16bf;
typedef __attribute__((ext_vector_type(8)))  __bf16 v8bf;
typedef __attribute__((ext_vector_type(8)))  float  v8f;

__device__ __forceinline__ int edge_sel(float d) {
  int s = (int)floorf((d - SEL_LO) / SEL_DIV * 8.0f);
  return (s >= 0 && s < NEXP) ? s : NEXP;     // out-of-range -> passthrough bucket
}

// ---- 16x128 (A, LDS bf16) x 128x16 (B^T row-major, bf16) -> 16x16 f32, one wave ----
// A fragment: lane holds row (lane&15); K chunks {32ks+8hi..+7} and {32ks+16+8hi..+7}
// B fragment: lane holds col (lane&15); K chunks {32ks+8hi..+7} and {32ks+16+8hi..+7} of B^T row
__device__ __forceinline__ v8f gemm16(const bf16_t* A, const bf16_t* BtRow, int hi, int rowA) {
  v8f c = {0.f, 0.f, 0.f, 0.f, 0.f, 0.f, 0.f, 0.f};
#pragma unroll
  for (int ks = 0; ks < 4; ++ks) {
    v8bf a0 = *(const v8bf*)&A[rowA * HP + ks * 32 + hi * 8];
    v8bf a1 = *(const v8bf*)&A[rowA * HP + ks * 32 + 16 + hi * 8];
    v16bf av = __builtin_shufflevector(a0, a1, 0,1,2,3,4,5,6,7,8,9,10,11,12,13,14,15);
    v8bf b0 = *(const v8bf*)&BtRow[ks * 32 + hi * 8];
    v8bf b1 = *(const v8bf*)&BtRow[ks * 32 + 16 + hi * 8];
    v16bf bv = __builtin_shufflevector(b0, b1, 0,1,2,3,4,5,6,7,8,9,10,11,12,13,14,15);
    c = __builtin_amdgcn_wmma_f32_16x16x32_bf16(false, av, false, bv, (short)0, c, false, false);
  }
  return c;
}

// ---------------- small prep kernels ----------------
__global__ void k_zero(float* acc, int* cnt, long n) {
  long idx = (long)blockIdx.x * 256 + threadIdx.x;
  for (long p = idx; p < n; p += (long)gridDim.x * 256) acc[p] = 0.0f;
  if (idx < NBKT) cnt[idx] = 0;
}

__global__ void k_prep_weights(const float* sel_w, const float* w1, const float* w2,
                               const float* w3, const float* w_out,
                               bf16_t* selT, bf16_t* wcat) {
  int idx = blockIdx.x * 256 + threadIdx.x;
  const int SHH = NEXP * H * H;
  if (idx < SHH) {
    int s = idx >> 14, rem = idx & 16383;
    int n = rem >> 7, k = rem & 127;
    selT[idx] = (bf16_t)sel_w[s * H * H + k * H + n];  // transpose: B^T[n][k] = B[k][n]
  } else {
    int o = idx - SHH;
    if (o < 4 * H * H) {
      int l = o >> 14, rem = o & 16383;
      const float* src = (l == 0) ? w1 : (l == 1) ? w2 : (l == 2) ? w3 : w_out;
      wcat[o] = (bf16_t)src[rem];                      // torch [out,in] layout IS B^T
    }
  }
}

__global__ void k_count(const float* dist, int* cnt, int E) {
  int e = blockIdx.x * 256 + threadIdx.x;
  if (e < E) atomicAdd(&cnt[edge_sel(dist[e])], 1);
}

__global__ void k_scan(const int* cnt, int* boff, int* cur) {
  int o = 0;
  boff[0] = 0;
  for (int b = 0; b < NBKT; ++b) {
    cur[b] = o;
    o += (cnt[b] + 15) & ~15;      // 16-align every bucket so tiles never straddle
    boff[b + 1] = o;
  }
}

__global__ void k_scatter(const float* dist, int* cur, int* perm, int E) {
  int e = blockIdx.x * 256 + threadIdx.x;
  if (e < E) {
    int p = atomicAdd(&cur[edge_sel(dist[e])], 1);
    perm[p] = e;
  }
}

// ---------------- edge stage: gate + expert WMMA GEMM + branch-free scatter ----------------
__global__ void __launch_bounds__(256)
k_edge(const float* x, const float* rbf, const int* ei,
       const int* perm, const int* cnt, const int* boff,
       const float* w_rbf, const bf16_t* selT, float* acc, long trashOff) {
  __shared__ __align__(16) bf16_t hS[16 * HP];
  __shared__ int  eIdx[16];
  __shared__ long rowOff[16];                 // element offset into acc (or trash)
  __shared__ float wrbfS[H * NRAD];           // 128x6 gate weights
  __shared__ float rbfS[16 * NRAD];           // 16 edges' radial features

  int tile_base = blockIdx.x * 16;
  if (tile_base >= boff[NBKT]) return;
  int b = 0;
#pragma unroll
  for (int q = 1; q < NBKT; ++q) if (tile_base >= boff[q]) b = q;
  int valid_end = boff[b] + cnt[b];

  int t = threadIdx.x;
  if (t < 16) {
    int p = tile_base + t;
    if (p < valid_end) {
      int e = perm[p];
      eIdx[t] = e;
      rowOff[t] = (long)ei[e] * H;
    } else {
      eIdx[t] = -1;
      rowOff[t] = trashOff;                   // zero rows dump harmlessly into trash
    }
  }
  for (int idx = t; idx < H * NRAD; idx += 256) wrbfS[idx] = w_rbf[idx];
  __syncthreads();

  if (t < 16 * NRAD) {
    int r = t / NRAD, j = t - r * NRAD;
    int e = eIdx[r];
    rbfS[t] = (e >= 0) ? rbf[(long)e * NRAD + j] : 0.f;
  }
  __syncthreads();

  // h = (rbf @ w_rbf^T) * x   -> bf16 tile in LDS (16 edges x 128)
  int r = t >> 4, kb = (t & 15) * 8;
  int e = eIdx[r];
  v8bf hv;
  if (e >= 0) {
    float4 x0 = *(const float4*)&x[(long)e * H + kb];
    float4 x1 = *(const float4*)&x[(long)e * H + kb + 4];
    float xs[8] = {x0.x, x0.y, x0.z, x0.w, x1.x, x1.y, x1.z, x1.w};
#pragma unroll
    for (int k = 0; k < 8; ++k) {
      int kk = kb + k;
      float g = 0.f;
#pragma unroll
      for (int j = 0; j < NRAD; ++j) g += wrbfS[kk * NRAD + j] * rbfS[r * NRAD + j];
      hv[k] = (bf16_t)(g * xs[k]);
    }
  } else {
#pragma unroll
    for (int k = 0; k < 8; ++k) hv[k] = (bf16_t)0.0f;
  }
  *(v8bf*)&hS[r * HP + kb] = hv;
  __syncthreads();

  if (b < NEXP) {
    int wave = t >> 5, lane = t & 31;
    int hi = lane >> 4, lo = lane & 15;
    int col = wave * 16 + lo;
    const bf16_t* BtRow = selT + (long)b * H * H + (long)col * H;
    v8f c = gemm16(hS, BtRow, hi, lo);
#pragma unroll
    for (int v = 0; v < 8; ++v) {
      int M = v + hi * 8;
      atomicAdd(&acc[rowOff[M] + col], c[v]);   // branch-free: invalid rows -> trash
    }
  } else {
    // passthrough bucket: out = h (zeros go to trash)
    long base = rowOff[r];
#pragma unroll
    for (int k = 0; k < 8; ++k)
      atomicAdd(&acc[base + kb + k], (float)hv[k]);
  }
}

// ---------------- node stage: fused 4-layer MLP, WMMA, LDS ping-pong ----------------
__global__ void __launch_bounds__(256)
k_mlp(const float* acc, const bf16_t* wcat, const float* b1, const float* b2,
      const float* b3, float* out, int N) {
  __shared__ __align__(16) bf16_t bufA[16 * HP];
  __shared__ __align__(16) bf16_t bufB[16 * HP];

  int t = threadIdx.x;
  int tile = blockIdx.x;
  int r = t >> 4, kb = (t & 15) * 8;
  long node = (long)tile * 16 + r;
  {
    v8bf hv;
    if (node < N) {
      float4 a0 = *(const float4*)&acc[node * H + kb];
      float4 a1 = *(const float4*)&acc[node * H + kb + 4];
      hv[0] = (bf16_t)a0.x; hv[1] = (bf16_t)a0.y; hv[2] = (bf16_t)a0.z; hv[3] = (bf16_t)a0.w;
      hv[4] = (bf16_t)a1.x; hv[5] = (bf16_t)a1.y; hv[6] = (bf16_t)a1.z; hv[7] = (bf16_t)a1.w;
    } else {
#pragma unroll
      for (int k = 0; k < 8; ++k) hv[k] = (bf16_t)0.0f;
    }
    *(v8bf*)&bufA[r * HP + kb] = hv;
  }
  __syncthreads();

  int wave = t >> 5, lane = t & 31;
  int hi = lane >> 4, lo = lane & 15;
  int col = wave * 16 + lo;

  const float* biases[3] = {b1, b2, b3};
#pragma unroll
  for (int l = 0; l < 3; ++l) {
    const bf16_t* A = (l & 1) ? bufB : bufA;
    bf16_t* Ob = (l & 1) ? bufA : bufB;
    const bf16_t* BtRow = wcat + (long)l * H * H + (long)col * H;
    v8f c = gemm16(A, BtRow, hi, lo);
    float bias = biases[l][col];
#pragma unroll
    for (int v = 0; v < 8; ++v) {
      int M = v + hi * 8;
      float z = c[v] + bias;
      float sg = 1.0f / (1.0f + __expf(-z));
      Ob[M * HP + col] = (bf16_t)(z * sg);
    }
    __syncthreads();
  }
  {
    const bf16_t* BtRow = wcat + 3L * H * H + (long)col * H;
    v8f c = gemm16(bufB, BtRow, hi, lo);   // layer3 output lives in bufB
#pragma unroll
    for (int v = 0; v < 8; ++v) {
      int M = v + hi * 8;
      long gn = (long)tile * 16 + M;
      if (gn < N) out[gn * OUT_F + col] = c[v];
    }
  }
}

// ---------------- host launcher ----------------
extern "C" void kernel_launch(void* const* d_in, const int* in_sizes, int n_in,
                              void* d_out, int out_size, void* d_ws, size_t ws_size,
                              hipStream_t stream) {
  const float* x     = (const float*)d_in[0];
  const float* rbf   = (const float*)d_in[1];
  const int*   ei    = (const int*)d_in[2];
  const float* dist  = (const float*)d_in[3];
  // d_in[4] = num_nodes scalar (N derived from out_size instead)
  const float* w_rbf = (const float*)d_in[5];
  const float* sel_w = (const float*)d_in[6];
  const float* w1    = (const float*)d_in[7];
  const float* b1    = (const float*)d_in[8];
  const float* w2    = (const float*)d_in[9];
  const float* b2    = (const float*)d_in[10];
  const float* w3    = (const float*)d_in[11];
  const float* b3    = (const float*)d_in[12];
  const float* w_out = (const float*)d_in[13];
  float* out = (float*)d_out;

  const int E = in_sizes[2];          // edge count (size of i)
  const int N = out_size / OUT_F;     // node count

  // carve workspace (256B aligned blocks)
  char* ws = (char*)d_ws;
  size_t o = 0;
  auto alloc = [&](size_t bytes) -> void* {
    o = (o + 255) & ~(size_t)255;
    void* p = ws + o;
    o += bytes;
    return p;
  };
  float*  acc   = (float*)alloc((size_t)N * H * sizeof(float));
  float*  trash = (float*)alloc(256 * sizeof(float));
  int*    perm  = (int*)alloc(((size_t)E + 16 * NBKT) * sizeof(int));
  int*    cnt   = (int*)alloc(NBKT * sizeof(int));
  int*    boff  = (int*)alloc((NBKT + 1) * sizeof(int));
  int*    cur   = (int*)alloc(NBKT * sizeof(int));
  bf16_t* selT  = (bf16_t*)alloc((size_t)NEXP * H * H * sizeof(bf16_t));
  bf16_t* wcat  = (bf16_t*)alloc((size_t)4 * H * H * sizeof(bf16_t));
  (void)ws_size; (void)n_in;
  long trashOff = (long)(trash - acc);   // element offset of trash relative to acc

  long accN = (long)N * H;
  int zb = (int)((accN + 255) / 256);
  k_zero<<<zb, 256, 0, stream>>>(acc, cnt, accN);

  int pw = (NEXP * H * H + 4 * H * H + 255) / 256;
  k_prep_weights<<<pw, 256, 0, stream>>>(sel_w, w1, w2, w3, w_out, selT, wcat);

  int eb = (E + 255) / 256;
  k_count<<<eb, 256, 0, stream>>>(dist, cnt, E);
  k_scan<<<1, 1, 0, stream>>>(cnt, boff, cur);
  k_scatter<<<eb, 256, 0, stream>>>(dist, cur, perm, E);

  int tiles = (E + 15) / 16 + NBKT;   // padded bucket tiles; extras early-exit
  k_edge<<<tiles, 256, 0, stream>>>(x, rbf, ei, perm, cnt, boff, w_rbf, selT, acc, trashOff);

  int nt = (N + 15) / 16;
  k_mlp<<<nt, 256, 0, stream>>>(acc, wcat, b1, b2, b3, out, N);
}
